// fwd_gnn_resid_45174466019871
// MI455X (gfx1250) — compile-verified, matrix operand order checked by reading, not run on hardware
//
#include <hip/hip_runtime.h>
#include <hip/hip_bf16.h>

typedef __attribute__((ext_vector_type(16))) _Float16 v16h;
typedef __attribute__((ext_vector_type(8)))  _Float16 v8h;
typedef __attribute__((ext_vector_type(4)))  _Float16 v4h;
typedef __attribute__((ext_vector_type(8)))  float    v8f;
typedef __attribute__((ext_vector_type(4)))  float    v4f;

#define TPB 128   // 4 wave32s per block
#define MBLK 64   // rows per block (16 rows per wave)

// ---------------------------------------------------------------------------
// gfx1250 hardware tanh (confirmed: lowers to v_tanh_f32). Branch-free
// fallback for older toolchains.
// ---------------------------------------------------------------------------
__device__ __forceinline__ float fast_tanh(float x) {
#if __has_builtin(__builtin_amdgcn_tanhf)
  return __builtin_amdgcn_tanhf(x);
#else
  float e = __builtin_amdgcn_exp2f(x * 2.8853900817779268f);
  float r = __builtin_amdgcn_rcpf(e + 1.0f);
  return 1.0f - 2.0f * r;
#endif
}

// ---------------------------------------------------------------------------
// Stage chain weights (fp32 global -> f16 LDS) directly in B-fragment layout:
// frag = 512 halves, lane-major; lane n holds column nBase+(n&15), halves
// j=0..7 -> K = kBase+kTop+j, j=8..15 -> K = kBase+16+kTop+(j-8),
// kTop = (lane>=16) ? 8 : 0.  (mirror of the ISA 16-bit A layout)
// ---------------------------------------------------------------------------
template <int KIN>
__device__ __forceinline__ void stage_weights(const float* __restrict__ W0,
                                              const float* __restrict__ b0,
                                              const float* __restrict__ Ws,
                                              const float* __restrict__ bs,
                                              _Float16* wfrag, float* bias) {
  const int k0tiles = KIN / 32;
  const int nfrag   = k0tiles * 4 + 72;
  for (int idx = threadIdx.x; idx < nfrag * 512; idx += blockDim.x) {
    int frag = idx >> 9;
    int r    = idx & 511;
    int lane = r >> 4;
    int j    = r & 15;
    int li, kt, nt;
    if (frag < k0tiles * 4) { li = 0; kt = frag >> 2; nt = frag & 3; }
    else { int f = frag - k0tiles * 4; li = 1 + (f >> 3); kt = (f >> 2) & 1; nt = f & 3; }
    int n    = nt * 16 + (lane & 15);
    int kTop = (lane >> 4) << 3;
    int k    = kt * 32 + ((j < 8) ? (kTop + j) : (16 + kTop + (j - 8)));
    float w  = (li == 0) ? W0[k * 64 + n] : Ws[(li - 1) * 4096 + k * 64 + n];
    wfrag[idx] = (_Float16)w;
  }
  for (int t = threadIdx.x; t < 640; t += blockDim.x)
    bias[t] = (t < 64) ? b0[t] : bs[t - 64];
}

// ---------------------------------------------------------------------------
// 10-layer Linear+Tanh chain with residual adds (layers 3,6,9) on 64 rows.
// Each wave owns rows [wave*16, wave*16+16) -> no intra-chain barriers.
// Final activations land in actB.
// ---------------------------------------------------------------------------
template <int KIN>
__device__ void run_chain(const _Float16* wfrag, const float* bias,
                          _Float16* actIn, _Float16* actA, _Float16* actB,
                          _Float16* resBuf) {
  const int lane = threadIdx.x & 31;
  const int wave = threadIdx.x >> 5;
  const int row0 = wave * 16;
  const int colN = lane & 15;
  const int rowC = (lane >> 4) << 3;
  const int kTop = (lane >> 4) << 3;

  const int k0tiles = KIN / 32;
  _Float16* src = actIn;
  int srcStride = KIN;
  _Float16* dst = actA;

#pragma unroll
  for (int li = 0; li < 10; ++li) {
    const int  kTiles   = (li == 0) ? k0tiles : 2;
    const int  fragBase = (li == 0) ? 0 : (k0tiles * 4 + (li - 1) * 8);
    const bool addRes   = (li == 3) || (li == 6) || (li == 9);
    const bool saveRes  = (li == 0) || (li == 3) || (li == 6);

    v8f acc0 = {}, acc1 = {}, acc2 = {}, acc3 = {};
#pragma unroll
    for (int kt = 0; kt < kTiles; ++kt) {
      const _Float16* p = src + (row0 + colN) * srcStride + kt * 32 + kTop;
      v8h lo = *(const v8h*)(p);
      v8h hi = *(const v8h*)(p + 16);
      if (addRes) {
        const _Float16* pr = resBuf + (row0 + colN) * 64 + kt * 32 + kTop;
        lo += *(const v8h*)(pr);
        hi += *(const v8h*)(pr + 16);
      }
      v16h a = __builtin_shufflevector(lo, hi, 0, 1, 2, 3, 4, 5, 6, 7,
                                       8, 9, 10, 11, 12, 13, 14, 15);
      const _Float16* wb = wfrag + (size_t)(fragBase + kt * 4) * 512 + lane * 16;
      v16h b0 = *(const v16h*)(wb);
      v16h b1 = *(const v16h*)(wb + 512);
      v16h b2 = *(const v16h*)(wb + 1024);
      v16h b3 = *(const v16h*)(wb + 1536);
      acc0 = __builtin_amdgcn_wmma_f32_16x16x32_f16(false, a, false, b0, (short)0, acc0, false, false);
      acc1 = __builtin_amdgcn_wmma_f32_16x16x32_f16(false, a, false, b1, (short)0, acc1, false, false);
      acc2 = __builtin_amdgcn_wmma_f32_16x16x32_f16(false, a, false, b2, (short)0, acc2, false, false);
      acc3 = __builtin_amdgcn_wmma_f32_16x16x32_f16(false, a, false, b3, (short)0, acc3, false, false);
    }
    const float* bl = bias + li * 64;
#pragma unroll
    for (int nt = 0; nt < 4; ++nt) {
      v8f acc = (nt == 0) ? acc0 : (nt == 1) ? acc1 : (nt == 2) ? acc2 : acc3;
      float bn = bl[nt * 16 + colN];
#pragma unroll
      for (int v = 0; v < 8; ++v) {
        float   x = fast_tanh(acc[v] + bn);
        _Float16 h = (_Float16)x;
        int r = row0 + rowC + v;
        int c = nt * 16 + colN;
        dst[r * 64 + c] = h;
        if (saveRes) resBuf[r * 64 + c] = h;
      }
    }
    src = dst; srcStride = 64;
    dst = (dst == actA) ? actB : actA;
  }
}

// ---------------------------------------------------------------------------
// embeds = tanh(node_feats @ W_embed + b_embed)   (K=32, one K-tile)
// ---------------------------------------------------------------------------
__global__ void embed_kernel(const float* __restrict__ feats,
                             const float* __restrict__ W,
                             const float* __restrict__ b,
                             float* __restrict__ out, int nrows) {
  extern __shared__ char smem[];
  _Float16* wfrag = (_Float16*)smem;
  _Float16* actIn = wfrag + 4 * 512;             // 64 x 32
  float*    bias  = (float*)(actIn + 64 * 32);

  for (int idx = threadIdx.x; idx < 4 * 512; idx += blockDim.x) {
    int frag = idx >> 9, r = idx & 511, lane = r >> 4, j = r & 15;
    int n = frag * 16 + (lane & 15);
    int kTop = (lane >> 4) << 3;
    int k = (j < 8) ? (kTop + j) : (16 + kTop + (j - 8));
    wfrag[idx] = (_Float16)W[k * 64 + n];
  }
  if (threadIdx.x < 64) bias[threadIdx.x] = b[threadIdx.x];

  const int  m0   = blockIdx.x * MBLK;
  const bool full = (m0 + MBLK) <= nrows;

  // stage 64x32 fp32 -> f16 (float4 loads, 8B LDS stores)
  for (int idx = threadIdx.x; idx < 64 * 8; idx += blockDim.x) {
    int row = idx >> 3, c4 = (idx & 7) << 2;
    int g = m0 + row;
    int gg = full ? g : (g < nrows ? g : nrows - 1);
    v4f x = *(const v4f*)(feats + (size_t)gg * 32 + c4);
    *(v4h*)(actIn + row * 32 + c4) = __builtin_convertvector(x, v4h);
  }
  __syncthreads();

  const int lane = threadIdx.x & 31;
  const int wave = threadIdx.x >> 5;
  const int row0 = wave * 16;
  const int colN = lane & 15;
  const int rowC = (lane >> 4) << 3;
  const int kTop = (lane >> 4) << 3;

  const _Float16* p = actIn + (row0 + colN) * 32 + kTop;
  v8h lo = *(const v8h*)(p);
  v8h hi = *(const v8h*)(p + 16);
  v16h a = __builtin_shufflevector(lo, hi, 0, 1, 2, 3, 4, 5, 6, 7,
                                   8, 9, 10, 11, 12, 13, 14, 15);
  v8f acc[4];
#pragma unroll
  for (int nt = 0; nt < 4; ++nt) {
    v16h bfr = *(const v16h*)(wfrag + nt * 512 + lane * 16);
    v8f z = {};
    acc[nt] = __builtin_amdgcn_wmma_f32_16x16x32_f16(false, a, false, bfr, (short)0, z, false, false);
  }
  if (full) {
#pragma unroll
    for (int nt = 0; nt < 4; ++nt) {
      float bn = bias[nt * 16 + colN];
#pragma unroll
      for (int v = 0; v < 8; ++v)
        out[(size_t)(m0 + row0 + rowC + v) * 64 + nt * 16 + colN] =
            fast_tanh(acc[nt][v] + bn);
    }
  } else {
#pragma unroll
    for (int nt = 0; nt < 4; ++nt) {
      float bn = bias[nt * 16 + colN];
#pragma unroll
      for (int v = 0; v < 8; ++v) {
        int r = m0 + row0 + rowC + v;
        if (r < nrows)
          out[(size_t)r * 64 + nt * 16 + colN] = fast_tanh(acc[nt][v] + bn);
      }
    }
  }
}

// ---------------------------------------------------------------------------
// Unary redux chain: gather 64-wide rows from `out`, run chain, store f16 redux
// ---------------------------------------------------------------------------
__global__ void uni_kernel(const float* __restrict__ out,
                           const int* __restrict__ preds, int off,
                           const float* __restrict__ W0, const float* __restrict__ b0,
                           const float* __restrict__ Ws, const float* __restrict__ bs,
                           _Float16* __restrict__ redux, int nrows) {
  extern __shared__ char smem[];
  const int NFRAG = 80;
  _Float16* wfrag = (_Float16*)smem;
  _Float16* actIn = wfrag + NFRAG * 512;
  _Float16* actA  = actIn + 64 * 64;
  _Float16* actB  = actA + 64 * 64;
  _Float16* resB  = actB + 64 * 64;
  float*    bias  = (float*)(resB + 64 * 64);

  stage_weights<64>(W0, b0, Ws, bs, wfrag, bias);

  const int  m0   = blockIdx.x * MBLK;
  const bool full = (m0 + MBLK) <= nrows;

  for (int idx = threadIdx.x; idx < 64 * 16; idx += blockDim.x) {
    int row = idx >> 4, c4 = (idx & 15) << 2;
    int g = m0 + row;
    int gg = full ? g : (g < nrows ? g : nrows - 1);
    int src = preds[gg] + off;
    v4f x = *(const v4f*)(out + (size_t)src * 64 + c4);
    *(v4h*)(actIn + row * 64 + c4) = __builtin_convertvector(x, v4h);
  }
  __syncthreads();
  run_chain<64>(wfrag, bias, actIn, actA, actB, resB);
  __syncthreads();
  if (full) {
    for (int idx = threadIdx.x; idx < 64 * 16; idx += blockDim.x) {
      int row = idx >> 4, c4 = (idx & 15) << 2;
      *(v4h*)(redux + (size_t)(m0 + row) * 64 + c4) =
          *(const v4h*)(actB + row * 64 + c4);
    }
  } else {
    for (int idx = threadIdx.x; idx < 64 * 16; idx += blockDim.x) {
      int row = idx >> 4, c4 = (idx & 15) << 2;
      int g = m0 + row;
      if (g < nrows)
        *(v4h*)(redux + (size_t)g * 64 + c4) = *(const v4h*)(actB + row * 64 + c4);
    }
  }
}

// ---------------------------------------------------------------------------
// Binary redux chain: gather two nodes -> 128-wide input
// ---------------------------------------------------------------------------
__global__ void bin_kernel(const float* __restrict__ out,
                           const int* __restrict__ preds, int off,
                           const float* __restrict__ W0, const float* __restrict__ b0,
                           const float* __restrict__ Ws, const float* __restrict__ bs,
                           _Float16* __restrict__ redux, int nrows) {
  extern __shared__ char smem[];
  const int NFRAG = 88;
  _Float16* wfrag = (_Float16*)smem;
  _Float16* actIn = wfrag + NFRAG * 512;       // 64 x 128
  _Float16* actA  = actIn + 64 * 128;
  _Float16* actB  = actA + 64 * 64;
  _Float16* resB  = actB + 64 * 64;
  float*    bias  = (float*)(resB + 64 * 64);

  stage_weights<128>(W0, b0, Ws, bs, wfrag, bias);

  const int  m0   = blockIdx.x * MBLK;
  const bool full = (m0 + MBLK) <= nrows;

  for (int idx = threadIdx.x; idx < 64 * 32; idx += blockDim.x) {
    int row = idx >> 5, c4 = (idx & 31) << 2;
    int g = m0 + row;
    int gg = full ? g : (g < nrows ? g : nrows - 1);
    int src = preds[gg * 2 + (c4 >> 6)] + off;
    v4f x = *(const v4f*)(out + (size_t)src * 64 + (c4 & 63));
    *(v4h*)(actIn + row * 128 + c4) = __builtin_convertvector(x, v4h);
  }
  __syncthreads();
  run_chain<128>(wfrag, bias, actIn, actA, actB, resB);
  __syncthreads();
  if (full) {
    for (int idx = threadIdx.x; idx < 64 * 16; idx += blockDim.x) {
      int row = idx >> 4, c4 = (idx & 15) << 2;
      *(v4h*)(redux + (size_t)(m0 + row) * 64 + c4) =
          *(const v4h*)(actB + row * 64 + c4);
    }
  } else {
    for (int idx = threadIdx.x; idx < 64 * 16; idx += blockDim.x) {
      int row = idx >> 4, c4 = (idx & 15) << 2;
      int g = m0 + row;
      if (g < nrows)
        *(v4h*)(redux + (size_t)g * 64 + c4) = *(const v4h*)(actB + row * 64 + c4);
    }
  }
}

// ---------------------------------------------------------------------------
// Node chain: input = [embeds(seg) , redux] (128-wide), writes out segment fp32
// ---------------------------------------------------------------------------
__global__ void node_kernel(float* __restrict__ out,
                            const _Float16* __restrict__ redux, int s,
                            const float* __restrict__ W0, const float* __restrict__ b0,
                            const float* __restrict__ Ws, const float* __restrict__ bs,
                            int nrows) {
  extern __shared__ char smem[];
  const int NFRAG = 88;
  _Float16* wfrag = (_Float16*)smem;
  _Float16* actIn = wfrag + NFRAG * 512;
  _Float16* actA  = actIn + 64 * 128;
  _Float16* actB  = actA + 64 * 64;
  _Float16* resB  = actB + 64 * 64;
  float*    bias  = (float*)(resB + 64 * 64);

  stage_weights<128>(W0, b0, Ws, bs, wfrag, bias);

  const int  m0   = blockIdx.x * MBLK;
  const bool full = (m0 + MBLK) <= nrows;

  // first 64 cols: embeds segment (fp32 -> f16)
  for (int idx = threadIdx.x; idx < 64 * 16; idx += blockDim.x) {
    int row = idx >> 4, c4 = (idx & 15) << 2;
    int g = m0 + row;
    int gg = full ? g : (g < nrows ? g : nrows - 1);
    v4f x = *(const v4f*)(out + (size_t)(s + gg) * 64 + c4);
    *(v4h*)(actIn + row * 128 + c4) = __builtin_convertvector(x, v4h);
  }
  // last 64 cols: redux (straight f16 copy)
  for (int idx = threadIdx.x; idx < 64 * 16; idx += blockDim.x) {
    int row = idx >> 4, c4 = (idx & 15) << 2;
    int g = m0 + row;
    int gg = full ? g : (g < nrows ? g : nrows - 1);
    *(v4h*)(actIn + row * 128 + 64 + c4) = *(const v4h*)(redux + (size_t)gg * 64 + c4);
  }
  __syncthreads();
  run_chain<128>(wfrag, bias, actIn, actA, actB, resB);
  __syncthreads();
  if (full) {
    for (int idx = threadIdx.x; idx < 64 * 16; idx += blockDim.x) {
      int row = idx >> 4, c4 = (idx & 15) << 2;
      v4h h = *(const v4h*)(actB + row * 64 + c4);
      *(v4f*)(out + (size_t)(s + m0 + row) * 64 + c4) = __builtin_convertvector(h, v4f);
    }
  } else {
    for (int idx = threadIdx.x; idx < 64 * 16; idx += blockDim.x) {
      int row = idx >> 4, c4 = (idx & 15) << 2;
      int g = m0 + row;
      if (g < nrows) {
        v4h h = *(const v4h*)(actB + row * 64 + c4);
        *(v4f*)(out + (size_t)(s + g) * 64 + c4) = __builtin_convertvector(h, v4f);
      }
    }
  }
}

// ---------------------------------------------------------------------------
extern "C" void kernel_launch(void* const* d_in, const int* in_sizes, int n_in,
                              void* d_out, int out_size, void* d_ws, size_t ws_size,
                              hipStream_t stream) {
  const float* node_feats = (const float*)d_in[0];
  const float* W_embed    = (const float*)d_in[1];
  const float* b_embed    = (const float*)d_in[2];
  const float* node_W0    = (const float*)d_in[3];
  const float* node_b0    = (const float*)d_in[4];
  const float* node_Ws    = (const float*)d_in[5];
  const float* node_bs    = (const float*)d_in[6];
  const float* uni_W0     = (const float*)d_in[7];
  const float* uni_b0     = (const float*)d_in[8];
  const float* uni_Ws     = (const float*)d_in[9];
  const float* uni_bs     = (const float*)d_in[10];
  const float* bin_W0     = (const float*)d_in[11];
  const float* bin_b0     = (const float*)d_in[12];
  const float* bin_Ws     = (const float*)d_in[13];
  const float* bin_bs     = (const float*)d_in[14];
  const int*   preds_u    = (const int*)d_in[15];
  const int*   preds_b    = (const int*)d_in[16];

  float*     out   = (float*)d_out;
  _Float16*  redux = (_Float16*)d_ws;   // PER x 64 f16 scratch

  const int NU  = in_sizes[15] / 9;
  const int NB  = in_sizes[16] / 18;
  const int PER = NU + NB;
  const int N   = in_sizes[0] / 32;

  const size_t shmE = (size_t)(4 * 512 + 64 * 32) * 2 + 64 * 4;
  const size_t shmU = (size_t)(80 * 512 + 64 * 64 * 4) * 2 + 640 * 4;
  const size_t shmB = (size_t)(88 * 512 + 64 * 128 + 64 * 64 * 3) * 2 + 640 * 4;

  hipFuncSetAttribute((const void*)embed_kernel, hipFuncAttributeMaxDynamicSharedMemorySize, (int)shmE);
  hipFuncSetAttribute((const void*)uni_kernel,   hipFuncAttributeMaxDynamicSharedMemorySize, (int)shmU);
  hipFuncSetAttribute((const void*)bin_kernel,   hipFuncAttributeMaxDynamicSharedMemorySize, (int)shmB);
  hipFuncSetAttribute((const void*)node_kernel,  hipFuncAttributeMaxDynamicSharedMemorySize, (int)shmB);

  embed_kernel<<<(N + MBLK - 1) / MBLK, TPB, shmE, stream>>>(node_feats, W_embed, b_embed, out, N);

  for (int l = 1; l <= 9; ++l) {
    const int off = (l - 1) * PER;
    const int s   = l * PER;
    uni_kernel<<<(NU + MBLK - 1) / MBLK, TPB, shmU, stream>>>(
        out, preds_u + (size_t)(l - 1) * NU, off,
        uni_W0, uni_b0, uni_Ws, uni_bs, redux, NU);
    bin_kernel<<<(NB + MBLK - 1) / MBLK, TPB, shmB, stream>>>(
        out, preds_b + (size_t)(l - 1) * NB * 2, off,
        bin_W0, bin_b0, bin_Ws, bin_bs, redux + (size_t)NU * 64, NB);
    node_kernel<<<(PER + MBLK - 1) / MBLK, TPB, shmB, stream>>>(
        out, redux, s, node_W0, node_b0, node_Ws, node_bs, PER);
  }
}
